// MultiheadAttention_38096359915769
// MI455X (gfx1250) — matrix operand order repaired and checked by reading, not compiled, not run on hardware
//
#include <hip/hip_runtime.h>

// MI455X / gfx1250, wave32. Flash attention with K/Q roles swapped per reference:
//   scores[k,q] = K[k,:]·Q[q,:] * 1/sqrt(hd); softmax over q; out[k,:] = P·V.
// Matmuls on v_wmma_f32_16x16x32_f16; Q/V tiles double-buffered in LDS via
// global_load_async_to_lds_b128 (ASYNCcnt), copies overlapped with compute.

#define Bc 2
#define Tc 4096
#define Dc 512
#define Hc 8
#define HDc 64

typedef __attribute__((ext_vector_type(16))) _Float16 v16h;
typedef __attribute__((ext_vector_type(8)))  float    v8f;

union Frag16 {
    v16h h;
    unsigned int u[8];
};

// 16 bytes per lane, global -> LDS, async (ASYNCcnt).
__device__ __forceinline__ void async_copy16(const _Float16* g, _Float16* l) {
    unsigned lds_off = (unsigned)(uintptr_t)l;  // low 32 bits = LDS byte offset
    asm volatile("global_load_async_to_lds_b128 %0, %1, off"
                 :: "v"(lds_off), "v"(g) : "memory");
}
__device__ __forceinline__ void async_wait_all() {
    asm volatile("s_wait_asynccnt 0" ::: "memory");
}
__device__ __forceinline__ void lds_wait_wave() {
    asm volatile("s_wait_dscnt 0" ::: "memory");
}

// ---------------------------------------------------------------------------
// Kernel 1: fused QKV projection. y[t,n] = sum_d x[t,d] * W[n,d] + b[n]
// blockIdx.z selects K / Q / V. Output f16:
//   K,Q -> [B,H,T,hd]; V -> [B,H,hd,T] (transposed for the PV WMMA B operand)
// ---------------------------------------------------------------------------
#define LDT 34  // LDS row stride in halfs

__global__ __launch_bounds__(128) void qkv_proj_kernel(
    const float* __restrict__ x,
    const float* __restrict__ Wk, const float* __restrict__ bk,
    const float* __restrict__ Wq, const float* __restrict__ bq,
    const float* __restrict__ Wv, const float* __restrict__ bv,
    _Float16* __restrict__ Kf, _Float16* __restrict__ Qf, _Float16* __restrict__ Vt)
{
    const int z = blockIdx.z;
    const float* __restrict__ W  = (z == 0) ? Wk : ((z == 1) ? Wq : Wv);
    const float* __restrict__ bb = (z == 0) ? bk : ((z == 1) ? bq : bv);
    _Float16* __restrict__ dst   = (z == 0) ? Kf : ((z == 1) ? Qf : Vt);

    __shared__ __align__(16) _Float16 As[64 * LDT];
    __shared__ __align__(16) _Float16 Bs[64 * LDT];

    const int tid  = threadIdx.x;
    const int lane = tid & 31;
    const int wid  = tid >> 5;
    const int lrow = lane & 15;
    const int half = lane >> 4;

    const int row0 = blockIdx.x * 64;  // flat token index b*T + t
    const int col0 = blockIdx.y * 64;  // output column n = h*hd + d

    v8f acc[4] = {};

    for (int k0 = 0; k0 < Dc; k0 += 32) {
        // statically unrolled staging: 4 loads in flight, then convert+store
        float4 va[4], vb[4];
#pragma unroll
        for (int it = 0; it < 4; ++it) {
            const int i = tid + it * 128;
            const int r = i >> 3, c4 = i & 7;
            va[it] = *(const float4*)(x + (size_t)(row0 + r) * Dc + k0 + c4 * 4);
            vb[it] = *(const float4*)(W + (size_t)(col0 + r) * Dc + k0 + c4 * 4);
        }
#pragma unroll
        for (int it = 0; it < 4; ++it) {
            const int i = tid + it * 128;
            const int r = i >> 3, c4 = i & 7;
            _Float16* pa = &As[r * LDT + c4 * 4];
            pa[0] = (_Float16)va[it].x; pa[1] = (_Float16)va[it].y;
            pa[2] = (_Float16)va[it].z; pa[3] = (_Float16)va[it].w;
            _Float16* pb = &Bs[r * LDT + c4 * 4];
            pb[0] = (_Float16)vb[it].x; pb[1] = (_Float16)vb[it].y;
            pb[2] = (_Float16)vb[it].z; pb[3] = (_Float16)vb[it].w;
        }
        __syncthreads();

        Frag16 a;
        const int ar = wid * 16 + lrow;
#pragma unroll
        for (int j = 0; j < 8; ++j) {
            const int kk = 2 * j + half * 16;
            a.u[j] = *(const unsigned int*)&As[ar * LDT + kk];
        }

#pragma unroll
        for (int cg = 0; cg < 4; ++cg) {
            Frag16 b;
            const int br = cg * 16 + lrow;
            const int kb = half * 16;
#pragma unroll
            for (int j = 0; j < 8; ++j)
                b.u[j] = *(const unsigned int*)&Bs[br * LDT + kb + 2 * j];
            acc[cg] = __builtin_amdgcn_wmma_f32_16x16x32_f16(
                false, a.h, false, b.h, (short)0, acc[cg], false, false);
        }
        __syncthreads();
    }

#pragma unroll
    for (int cg = 0; cg < 4; ++cg) {
        const int n = col0 + cg * 16 + lrow;
        const float bias = bb[n];
        const int h = n >> 6;
        const int d = n & (HDc - 1);
#pragma unroll
        for (int r = 0; r < 8; ++r) {
            const int bt = row0 + wid * 16 + r + half * 8;
            const int b  = bt >> 12;
            const int t  = bt & (Tc - 1);
            const float val = acc[cg][r] + bias;
            size_t off;
            if (z == 2)
                off = (((size_t)(b * Hc + h) * HDc + d) * Tc + t);   // Vt [B,H,hd,T]
            else
                off = (((size_t)(b * Hc + h) * Tc + t) * HDc + d);   // [B,H,T,hd]
            dst[off] = (_Float16)val;
        }
    }
}

// ---------------------------------------------------------------------------
// Kernel 2: flash attention. Block = 256 threads = 8 waves, all on the same
// (b,h); wave w owns k-tile rows [k0, k0+16). Per 64-wide q block:
//   - Q (64x64) and Vt (64x64) f16 tiles double-buffered in LDS; the next
//     block's async copies are issued right after the barrier so they overlap
//     the 16 WMMAs + softmax of the current block
// ---------------------------------------------------------------------------
#define STR  72   // Qs/Vs LDS row stride in halfs (144 B, 16B-aligned rows)
#define STRP 66   // P tile row stride in halfs

__device__ __forceinline__ void stage_tiles(
    const _Float16* __restrict__ Qg,   // -> Q rows [q0..q0+63] x 64 d
    const _Float16* __restrict__ Vg,   // -> Vt rows d=0..63, cols [q0..q0+63]
    _Float16* qs, _Float16* vs, int tid)
{
#pragma unroll
    for (int it = 0; it < 2; ++it) {
        const int i = tid + it * 256;
        const int r = i >> 3, c = (i & 7) * 8;
        async_copy16(Qg + (size_t)r * HDc + c, qs + r * STR + c);
    }
#pragma unroll
    for (int it = 0; it < 2; ++it) {
        const int i = tid + it * 256;
        const int r = i >> 3, c = (i & 7) * 8;
        async_copy16(Vg + (size_t)r * Tc + c, vs + r * STR + c);
    }
}

__global__ __launch_bounds__(256) void flash_attn_kernel(
    const _Float16* __restrict__ Kf, const _Float16* __restrict__ Qf,
    const _Float16* __restrict__ Vt, float* __restrict__ out)
{
    __shared__ __align__(16) _Float16 Qs[2][64 * STR];    // [buf][q][d]
    __shared__ __align__(16) _Float16 Vs[2][64 * STR];    // [buf][d][q]
    __shared__ __align__(16) _Float16 Pt[8][16 * STRP];   // per-wave [k][q]

    const int tid  = threadIdx.x;
    const int lane = tid & 31;
    const int wid  = tid >> 5;
    const int lrow = lane & 15;
    const int half = lane >> 4;

    const int gw = blockIdx.x * 8 + wid;     // global wave id
    const int bh = gw >> 8;                  // / (T/16) = /256, 0..15
    const int k0 = (gw & 255) * 16;

    const size_t baseKQ = (size_t)bh * Tc * HDc;  // [bh][T][hd]
    const size_t baseV  = (size_t)bh * HDc * Tc;  // [bh][hd][T]

    // K rows as WMMA A operand (contract over d=64 -> two 16x32 fragments)
    Frag16 aK[2];
    {
        const _Float16* kp = Kf + baseKQ + (size_t)(k0 + lrow) * HDc;
#pragma unroll
        for (int s = 0; s < 2; ++s)
#pragma unroll
            for (int j = 0; j < 8; ++j) {
                const int kk = s * 32 + 2 * j + half * 16;
                aK[s].u[j] = *(const unsigned int*)(kp + kk);
            }
    }

    v8f o[4] = {};
    float m[8], l[8];
#pragma unroll
    for (int r = 0; r < 8; ++r) { m[r] = -1e30f; l[r] = 0.0f; }
    const float scale = 0.125f;  // 1/sqrt(64)

    // prologue: stage the first q block into buffer 0
    stage_tiles(Qf + baseKQ, Vt + baseV, &Qs[0][0], &Vs[0][0], tid);

    int cur = 0;
    for (int q0 = 0; q0 < Tc; q0 += 64, cur ^= 1) {
        // buf[cur] copies landed (all waves), and every wave is done reading
        // buf[cur^1] from the previous iteration -> safe to refill it.
        async_wait_all();
        __syncthreads();
        if (q0 + 64 < Tc)
            stage_tiles(Qf + baseKQ + (size_t)(q0 + 64) * HDc,
                        Vt + baseV + (q0 + 64),
                        &Qs[cur ^ 1][0], &Vs[cur ^ 1][0], tid);

        const _Float16* qbuf = &Qs[cur][0];
        const _Float16* vbuf = &Vs[cur][0];

        // ---- scores: four 16x16 tiles over this q block ----
        v8f sc[4] = {};
#pragma unroll
        for (int s = 0; s < 2; ++s) {
#pragma unroll
            for (int qg = 0; qg < 4; ++qg) {
                Frag16 b;
                const _Float16* qp = qbuf + (qg * 16 + lrow) * STR + s * 32 + half * 16;
                *(int4*)&b.u[0] = *(const int4*)(qp);
                *(int4*)&b.u[4] = *(const int4*)(qp + 8);
                sc[qg] = __builtin_amdgcn_wmma_f32_16x16x32_f16(
                    false, aK[s].h, false, b.h, (short)0, sc[qg], false, false);
            }
        }

        // ---- online softmax (row stats across the 16-lane half) ----
        float alpha[8];
#pragma unroll
        for (int r = 0; r < 8; ++r) {
            float mb = fmaxf(fmaxf(sc[0][r], sc[1][r]), fmaxf(sc[2][r], sc[3][r]));
            for (int mask = 1; mask < 16; mask <<= 1)
                mb = fmaxf(mb, __shfl_xor(mb, mask, 16));
            const float mn = fmaxf(m[r], mb);
            alpha[r] = __expf((m[r] - mn) * scale);
            float rs = 0.0f;
#pragma unroll
            for (int qg = 0; qg < 4; ++qg) {
                const float p = __expf((sc[qg][r] - mn) * scale);
                Pt[wid][(r + 8 * half) * STRP + qg * 16 + lrow] = (_Float16)p;
                rs += p;
            }
            for (int mask = 1; mask < 16; mask <<= 1)
                rs += __shfl_xor(rs, mask, 16);
            l[r] = l[r] * alpha[r] + rs;
            m[r] = mn;
        }
#pragma unroll
        for (int g = 0; g < 4; ++g)
#pragma unroll
            for (int r = 0; r < 8; ++r)
                o[g][r] *= alpha[r];

        // Pt is wave-private: same-wave DS ops are in-order; just fence.
        lds_wait_wave();

        // ---- PV: contract over the 64 q values (2 A-steps x 4 d-groups) ----
#pragma unroll
        for (int qs = 0; qs < 2; ++qs) {
            Frag16 aP;
#pragma unroll
            for (int j = 0; j < 8; ++j) {
                const int kk = qs * 32 + 2 * j + half * 16;
                aP.u[j] = *(const unsigned int*)&Pt[wid][lrow * STRP + kk];
            }
#pragma unroll
            for (int g = 0; g < 4; ++g) {
                Frag16 bV;
                const _Float16* vp = vbuf + (g * 16 + lrow) * STR + qs * 32 + half * 16;
                *(int4*)&bV.u[0] = *(const int4*)(vp);
                *(int4*)&bV.u[4] = *(const int4*)(vp + 8);
                o[g] = __builtin_amdgcn_wmma_f32_16x16x32_f16(
                    false, aP.h, false, bV.h, (short)0, o[g], false, false);
            }
        }
        // no end barrier: next iteration's top barrier fences buf reuse
    }

    // ---- normalize and write out[b, t=k, h, d] (f32) ----
    const int b = bh >> 3, h = bh & 7;
#pragma unroll
    for (int g = 0; g < 4; ++g) {
        const int d = g * 16 + lrow;
#pragma unroll
        for (int r = 0; r < 8; ++r) {
            const int krow = k0 + r + 8 * half;
            out[(size_t)(b * Tc + krow) * Dc + h * HDc + d] = o[g][r] / l[r];
        }
    }
}

// ---------------------------------------------------------------------------
extern "C" void kernel_launch(void* const* d_in, const int* in_sizes, int n_in,
                              void* d_out, int out_size, void* d_ws, size_t ws_size,
                              hipStream_t stream) {
    (void)in_sizes; (void)n_in; (void)out_size; (void)ws_size;
    const float* x  = (const float*)d_in[0];
    const float* Wk = (const float*)d_in[1];
    const float* bk = (const float*)d_in[2];
    const float* Wq = (const float*)d_in[3];
    const float* bq = (const float*)d_in[4];
    const float* Wv = (const float*)d_in[5];
    const float* bv = (const float*)d_in[6];
    float* out = (float*)d_out;

    const size_t nElem = (size_t)Bc * Hc * Tc * HDc;  // 4M f16 each
    _Float16* Kf = (_Float16*)d_ws;
    _Float16* Qf = Kf + nElem;
    _Float16* Vt = Qf + nElem;

    dim3 gProj((Bc * Tc) / 64, Dc / 64, 3);
    qkv_proj_kernel<<<gProj, 128, 0, stream>>>(x, Wk, bk, Wq, bq, Wv, bv, Kf, Qf, Vt);

    // B*H*(T/16) = 4096 k-tile waves, 8 waves (one (b,h)) per 256-thread block
    flash_attn_kernel<<<(Bc * Hc * (Tc / 16)) / 8, 256, 0, stream>>>(Kf, Qf, Vt, out);
}
